// ProteinFeaturesLigand_54357106098678
// MI455X (gfx1250) — matrix operand order repaired and checked by reading, not compiled, not run on hardware
//
#include <hip/hip_runtime.h>
#include <hip/hip_bf16.h>

// ---------------------------------------------------------------------------
// ProteinFeaturesLigand forward for MI455X (gfx1250, wave32, WMMA)
//
// Outputs (flat, in return order):
//   V        (2,1024,16,128)      -> 4,194,304 f32
//   E        (2,1024,30,128)      -> 7,864,320 f32
//   E_idx    (2,1024,30)          ->    61,440 (written as float values)
//   Y_nodes  (2,1024,16,128)      -> 4,194,304 f32
//   Y_edges  (2,1024,16,16,128)   -> 67,108,864 f32
//   Y_m      (2,1024,16)          ->    32,768 f32
// ---------------------------------------------------------------------------

typedef __attribute__((ext_vector_type(16))) _Float16 v16h;
typedef __attribute__((ext_vector_type(8)))  _Float16 v8h;
typedef __attribute__((ext_vector_type(8)))  float    v8f;
typedef __attribute__((ext_vector_type(4)))  float    v4f;

#define BDIM   2
#define LDIM   1024
#define KNBR   30
#define MPTS   16
#define NF     128

// ---- element tables (one-hot group/period gathers) ----
__constant__ int G_TBL[] = {0, 1, 18, 1, 2, 13, 14, 15, 16, 17, 18, 1, 2, 13, 14, 15, 16,
  17, 18, 1, 2, 3, 4, 5, 6, 7, 8, 9, 10, 11, 12, 13, 14, 15, 16, 17, 18, 1, 2, 3, 4, 5,
  6, 7, 8, 9, 10, 11, 12, 13, 14, 15, 16, 17, 18, 1, 2, 3, 3, 3, 3, 3, 3, 3, 3, 3, 3, 3,
  3, 3, 3, 3, 4, 5, 6, 7, 8, 9, 10, 11, 12, 13, 14, 15, 16, 17, 18, 1, 2, 3, 3, 3, 3, 3,
  3, 3, 3, 3, 3, 3, 3, 3, 3, 3, 4, 5, 6, 7, 8, 9, 10, 11, 12, 13, 14, 15, 16, 17, 18};
__constant__ int P_TBL[] = {0, 1, 1, 2, 2, 2, 2, 2, 2, 2, 2, 3, 3, 3, 3, 3, 3, 3, 3, 4,
  4, 4, 4, 4, 4, 4, 4, 4, 4, 4, 4, 4, 4, 4, 4, 4, 4, 5, 5, 5, 5, 5, 5, 5, 5, 5, 5, 5, 5,
  5, 5, 5, 5, 5, 5, 6, 6, 6, 6, 6, 6, 6, 6, 6, 6, 6, 6, 6, 6, 6, 6, 6, 6, 6, 6, 6, 6, 6,
  6, 6, 6, 6, 6, 6, 6, 6, 6, 7, 7, 7, 7, 7, 7, 7, 7, 7, 7, 7, 7, 7, 7, 7, 7, 7, 7, 7, 7,
  7, 7, 7, 7, 7, 7, 7, 7, 7, 7, 7, 7};

// 24 atom pairs (A from residue i, B from neighbor j); atoms: N=0,Ca=1,C=2,O=3,Cb=4
__constant__ int PAIR_A[24] = {0,2,3,4,1,1,1,1,0,0,0,4,4,3,0,2,3,4,2,3,4,2,3,2};
__constant__ int PAIR_B[24] = {0,2,3,4,0,2,3,4,2,3,4,2,3,2,1,1,1,1,0,0,0,4,4,3};

__device__ __forceinline__ float rbf_c(float d, int c) {
  float mu = 2.0f + (20.0f / 15.0f) * (float)c;   // linspace(2,22,16)
  float z  = (d - mu) * (1.0f / 1.25f);           // sigma = 20/16
  return __expf(-z * z);
}

// ---------------------------------------------------------------------------
// K0: per-residue atoms (N, Ca, C, O, Cb) -> ws
// ---------------------------------------------------------------------------
__global__ __launch_bounds__(256) void atoms_kernel(const float* __restrict__ X,
                                                    float* __restrict__ atoms, int total) {
  int t = blockIdx.x * blockDim.x + threadIdx.x;
  if (t >= total) return;
  const float* x = X + (long)t * 12;
  float Nx = x[0], Ny = x[1], Nz = x[2];
  float Ax = x[3], Ay = x[4], Az = x[5];
  float Cx = x[6], Cy = x[7], Cz = x[8];
  float Ox = x[9], Oy = x[10], Oz = x[11];
  float bx = Ax - Nx, by = Ay - Ny, bz = Az - Nz;
  float cx = Cx - Ax, cy = Cy - Ay, cz = Cz - Az;
  float ax = by * cz - bz * cy, ay = bz * cx - bx * cz, az = bx * cy - by * cx;
  float Bx = -0.58273431f * ax + 0.56802827f * bx - 0.54067466f * cx + Ax;
  float By = -0.58273431f * ay + 0.56802827f * by - 0.54067466f * cy + Ay;
  float Bz = -0.58273431f * az + 0.56802827f * bz - 0.54067466f * cz + Az;
  float* o = atoms + (long)t * 15;
  o[0] = Nx; o[1] = Ny; o[2] = Nz;
  o[3] = Ax; o[4] = Ay; o[5] = Az;
  o[6] = Cx; o[7] = Cy; o[8] = Cz;
  o[9] = Ox; o[10] = Oy; o[11] = Oz;
  o[12] = Bx; o[13] = By; o[14] = Bz;
}

// ---------------------------------------------------------------------------
// K1: masked pairwise Ca distances + top-30 selection. One block per (b,i).
// ---------------------------------------------------------------------------
__global__ __launch_bounds__(256) void dist_topk_kernel(
    const float* __restrict__ X, const float* __restrict__ mask,
    float* __restrict__ dneigh, int* __restrict__ eidx, float* __restrict__ eidx_f) {
  __shared__ float Dsh[LDIM];
  __shared__ float Msh[LDIM];
  __shared__ float rval[256];
  __shared__ int   ridx[256];

  const int tid = threadIdx.x;
  const int b = blockIdx.x >> 10;
  const int i = blockIdx.x & (LDIM - 1);
  const long base = (long)b * LDIM;
  const float* Xi = X + ((base + i) * 4 + 1) * 3;
  const float cax = Xi[0], cay = Xi[1], caz = Xi[2];
  const float mi = mask[base + i];

  float lmax = -3.4e38f;
  for (int j = tid; j < LDIM; j += 256) {
    const float* Xj = X + ((base + j) * 4 + 1) * 3;
    float dx = cax - Xj[0], dy = cay - Xj[1], dz = caz - Xj[2];
    float m2 = mi * mask[base + j];
    float d  = m2 * sqrtf(dx * dx + dy * dy + dz * dz + 1e-6f);
    Dsh[j] = d; Msh[j] = m2;
    lmax = fmaxf(lmax, d);
  }
  rval[tid] = lmax;
  __syncthreads();
  for (int s = 128; s > 0; s >>= 1) {
    if (tid < s) rval[tid] = fmaxf(rval[tid], rval[tid + s]);
    __syncthreads();
  }
  const float dmax = rval[0];
  __syncthreads();
  for (int j = tid; j < LDIM; j += 256) Dsh[j] += (1.0f - Msh[j]) * dmax;
  __syncthreads();

  const long erow = (base + i) * (long)KNBR;
  for (int k = 0; k < KNBR; ++k) {
    float bv = 3.0e38f; int bj = -1;
    for (int j = tid; j < LDIM; j += 256) {
      float v = Dsh[j];
      if (v < bv || (v == bv && j < bj)) { bv = v; bj = j; }
    }
    rval[tid] = bv; ridx[tid] = bj;
    __syncthreads();
    for (int s = 128; s > 0; s >>= 1) {
      if (tid < s) {
        float ov = rval[tid + s]; int oj = ridx[tid + s];
        if (ov < rval[tid] || (ov == rval[tid] && oj < ridx[tid])) {
          rval[tid] = ov; ridx[tid] = oj;
        }
      }
      __syncthreads();
    }
    if (tid == 0) {
      int bj0 = ridx[0];
      dneigh[erow + k] = rval[0];
      eidx[erow + k]   = bj0;
      eidx_f[erow + k] = (float)bj0;
      Dsh[bj0] = 3.4e38f;       // exclude from next pass
    }
    __syncthreads();
  }
}

// ---------------------------------------------------------------------------
// K2: edge features -> f16 [nE, 416]: E_pos(16) | rbf(Dnbr)(16) | 24 pair rbfs
// One wave32 per edge, 8 edges per block; pair RBFs stored as 2x b128.
// ---------------------------------------------------------------------------
__global__ __launch_bounds__(256) void edge_feat_kernel(
    const float* __restrict__ atoms, const int* __restrict__ eidx,
    const float* __restrict__ dneigh, const int* __restrict__ R_idx,
    const int* __restrict__ chains, const float* __restrict__ pe_w,
    const float* __restrict__ pe_b, _Float16* __restrict__ feat, long nEdges) {
  long e = (long)blockIdx.x * 8 + (threadIdx.x >> 5);
  if (e >= nEdges) return;
  const int lane = threadIdx.x & 31;
  const long bi = e / KNBR;                 // b*L + i
  const int b  = (int)(bi / LDIM);
  const int j  = eidx[e];
  const long jb = (long)b * LDIM + j;
  _Float16* fr = feat + e * 416;

  if (lane < 16) {
    int off = R_idx[bi] - R_idx[jb];
    int ec  = (chains[bi] == chains[jb]) ? 1 : 0;
    int d   = off + 32; d = d < 0 ? 0 : (d > 64 ? 64 : d);
    int dpos = ec ? d : 65;
    fr[lane] = (_Float16)(pe_w[dpos * 16 + lane] + pe_b[lane]);
  } else {
    int c = lane - 16;
    fr[16 + c] = (_Float16)rbf_c(dneigh[e], c);
  }
  if (lane < 24) {
    const float* Ai = atoms + bi * 15 + PAIR_A[lane] * 3;
    const float* Aj = atoms + jb * 15 + PAIR_B[lane] * 3;
    float dx = Ai[0] - Aj[0], dy = Ai[1] - Aj[1], dz = Ai[2] - Aj[2];
    float d = sqrtf(dx * dx + dy * dy + dz * dz + 1e-6f);
    v8h h0, h1;
    #pragma unroll
    for (int c = 0; c < 8; ++c) h0[c] = (_Float16)rbf_c(d, c);
    #pragma unroll
    for (int c = 0; c < 8; ++c) h1[c] = (_Float16)rbf_c(d, 8 + c);
    _Float16* fp = fr + 32 + lane * 16;     // 32B aligned
    *(v8h*)fp       = h0;
    *(v8h*)(fp + 8) = h1;
  }
}

// ---------------------------------------------------------------------------
// K4: ligand-node features -> f16 [nP, 160]:
//   5x16 atom->Y RBFs | 64 Yt_emb (one-hot gather of tl_w rows) | 4 angle | pad
// One wave32 per point; each lane fills 5 feature slots.
// ---------------------------------------------------------------------------
__global__ __launch_bounds__(256) void node_feat_kernel(
    const float* __restrict__ atoms, const float* __restrict__ Y,
    const int* __restrict__ Y_t, const float* __restrict__ tl_w,
    const float* __restrict__ tl_b, _Float16* __restrict__ feat, long nPts) {
  long p = (long)blockIdx.x * 8 + (threadIdx.x >> 5);
  if (p >= nPts) return;
  const int lane = threadIdx.x & 31;
  const long bi = p / MPTS;
  const float* A = atoms + bi * 15;
  const float* y = Y + p * 3;
  const float yx = y[0], yy = y[1], yz = y[2];
  _Float16* fr = feat + p * 160;

  #pragma unroll
  for (int s = 0; s < 5; ++s) {
    int c = lane * 5 + s;       // 0..159
    float v = 0.0f;
    if (c < 80) {
      int a = c >> 4, cc = c & 15;
      float dx = A[a * 3 + 0] - yx, dy = A[a * 3 + 1] - yy, dz = A[a * 3 + 2] - yz;
      v = rbf_c(sqrtf(dx * dx + dy * dy + dz * dz + 1e-6f), cc);
    } else if (c < 144) {
      int cc = c - 80;
      int t = Y_t[p];
      int g = G_TBL[t], pd = P_TBL[t];
      v = tl_w[t * 64 + cc] + tl_w[(120 + g) * 64 + cc] + tl_w[(139 + pd) * 64 + cc] + tl_b[cc];
    } else if (c < 148) {
      // local-frame angle features
      float v1x = A[0] - A[3], v1y = A[1] - A[4], v1z = A[2] - A[5];
      float v2x = A[6] - A[3], v2y = A[7] - A[4], v2z = A[8] - A[5];
      float n1 = fmaxf(sqrtf(v1x * v1x + v1y * v1y + v1z * v1z), 1e-12f);
      float e1x = v1x / n1, e1y = v1y / n1, e1z = v1z / n1;
      float dp = e1x * v2x + e1y * v2y + e1z * v2z;
      float u2x = v2x - e1x * dp, u2y = v2y - e1y * dp, u2z = v2z - e1z * dp;
      float n2 = fmaxf(sqrtf(u2x * u2x + u2y * u2y + u2z * u2z), 1e-12f);
      float e2x = u2x / n2, e2y = u2y / n2, e2z = u2z / n2;
      float e3x = e1y * e2z - e1z * e2y;
      float e3y = e1z * e2x - e1x * e2z;
      float e3z = e1x * e2y - e1y * e2x;
      float dx = yx - A[3], dy = yy - A[4], dz = yz - A[5];
      float lx = e1x * dx + e1y * dy + e1z * dz;
      float ly = e2x * dx + e2y * dy + e2z * dz;
      float lz = e3x * dx + e3y * dy + e3z * dz;
      float rxy  = sqrtf(lx * lx + ly * ly + 1e-8f);
      float rxyz = sqrtf(lx * lx + ly * ly + lz * lz) + 1e-8f;
      int q = c - 144;
      v = (q == 0) ? lx / rxy : (q == 1) ? ly / rxy : (q == 2) ? rxy / rxyz : lz / rxyz;
    }
    fr[c] = (_Float16)v;
  }
}

// ---------------------------------------------------------------------------
// K5: Y-Y edge features -> f16 [nRows, 32] (16 RBF + 16 zero pad)
// ---------------------------------------------------------------------------
__global__ __launch_bounds__(256) void yedge_feat_kernel(
    const float* __restrict__ Y, _Float16* __restrict__ feat, long nRows) {
  long idx = (long)blockIdx.x * blockDim.x + threadIdx.x;
  if (idx >= nRows * 32) return;
  long row = idx >> 5;
  int c = (int)(idx & 31);
  float v = 0.0f;
  if (c < 16) {
    long p1 = row / MPTS;            // (b*L+i)*M + m1
    int m2  = (int)(row % MPTS);
    long bi = p1 / MPTS;
    const float* y1 = Y + p1 * 3;
    const float* y2 = Y + (bi * MPTS + m2) * 3;
    float dx = y1[0] - y2[0], dy = y1[1] - y2[1], dz = y1[2] - y2[2];
    v = rbf_c(sqrtf(dx * dx + dy * dy + dz * dz + 1e-6f), c);
  }
  feat[row * 32 + c] = (_Float16)v;
}

// ---------------------------------------------------------------------------
// Weight repack: f32 [Kdim,128] -> f16 fragment-major so each lane's WMMA B
// fragment is 32 contiguous bytes:
//   Wp[(((kt*8 + wv)*32 + lane)*16) + e] = W[kt*32 + (lane>>4)*16 + e][wv*16 + (lane&15)]
// ---------------------------------------------------------------------------
__global__ __launch_bounds__(256) void repack_w_kernel(
    const float* __restrict__ W, _Float16* __restrict__ Wp, int Kdim, int Kpad) {
  int idx = blockIdx.x * 256 + threadIdx.x;
  int total = (Kpad >> 5) * 8 * 32 * 16;
  if (idx >= total) return;
  int e    = idx & 15;
  int lane = (idx >> 4) & 31;
  int wv   = (idx >> 9) & 7;
  int kt   = idx >> 12;
  int k = kt * 32 + (lane >> 4) * 16 + e;
  int n = wv * 16 + (lane & 15);
  Wp[idx] = (k < Kdim) ? (_Float16)W[(long)k * 128 + n] : (_Float16)0.0f;
}

// ---------------------------------------------------------------------------
// WMMA GEMM + fused LayerNorm: out[rows,128] = LN(Af16[rows,KPAD] @ W + bias)
// Block = 256 threads = 8 wave32. Each wave owns one 16-col N-tile across a
// 32-row stripe (2 accumulators). A tile staged in LDS and read as 2x
// ds_load_b128 per fragment; B fragments read directly from repacked weights
// as 2x global_load_b128 (L2-resident). KPAD templated -> k-loop unrolled.
// ---------------------------------------------------------------------------
template <int KPAD>
__global__ __launch_bounds__(256) void wmma_gemm_ln_kernel(
    const _Float16* __restrict__ A, const _Float16* __restrict__ Wp,
    const float* __restrict__ bias, const float* __restrict__ gamma,
    const float* __restrict__ beta, float* __restrict__ out) {
  constexpr int KTILES = KPAD >> 5;
  __shared__ _Float16 Ah[32][40];     // 32x32 A tile, 80B row stride (16B aligned)
  __shared__ float    Cs[32][132];    // f32 out tile for LN
  __shared__ float    mu_s[32], rs_s[32];

  const int tid  = threadIdx.x;
  const int lane = tid & 31;
  const int wv   = tid >> 5;          // N-tile 0..7
  const int m    = lane & 15;
  const int hi   = lane >> 4;
  const long row0 = (long)blockIdx.x * 32;

  v8f acc0 = {}, acc1 = {};
  #pragma unroll
  for (int kt = 0; kt < KTILES; ++kt) {
    // stage A 32x32 halves cooperatively as b128 chunks
    for (int idx = tid; idx < 32 * 4; idx += 256) {
      int r = idx >> 2, cc = idx & 3;
      *(v8h*)&Ah[r][cc * 8] =
          *(const v8h*)&A[(row0 + r) * (long)KPAD + kt * 32 + cc * 8];
    }
    // B fragment: 32 contiguous bytes per lane from repacked weights
    const _Float16* bp = Wp + (((long)kt * 8 + wv) * 32 + lane) * 16;
    v8h b0 = *(const v8h*)bp;
    v8h b1 = *(const v8h*)(bp + 8);
    v16h bf = __builtin_shufflevector(b0, b1, 0, 1, 2, 3, 4, 5, 6, 7,
                                      8, 9, 10, 11, 12, 13, 14, 15);
    if (kt + 1 < KTILES)
      __builtin_prefetch(bp + (long)8 * 32 * 16, 0, 1);   // next k-tile of Wp
    __syncthreads();

    // A fragment: lane (m,hi) holds K = [hi*16, hi*16+16)
    v8h a0 = *(const v8h*)&Ah[m][hi * 16];
    v8h a1 = *(const v8h*)&Ah[m][hi * 16 + 8];
    v16h af0 = __builtin_shufflevector(a0, a1, 0, 1, 2, 3, 4, 5, 6, 7,
                                       8, 9, 10, 11, 12, 13, 14, 15);
    v8h a2 = *(const v8h*)&Ah[16 + m][hi * 16];
    v8h a3 = *(const v8h*)&Ah[16 + m][hi * 16 + 8];
    v16h af1 = __builtin_shufflevector(a2, a3, 0, 1, 2, 3, 4, 5, 6, 7,
                                       8, 9, 10, 11, 12, 13, 14, 15);

    acc0 = __builtin_amdgcn_wmma_f32_16x16x32_f16(false, af0, false, bf,
                                                  (short)0, acc0, false, false);
    acc1 = __builtin_amdgcn_wmma_f32_16x16x32_f16(false, af1, false, bf,
                                                  (short)0, acc1, false, false);
    __syncthreads();
  }

  // D layout: VGPR r -> row (hi*8 + r), col = wv*16 + (lane&15)
  const int col = wv * 16 + m;
  const float bcol = bias ? bias[col] : 0.0f;
  #pragma unroll
  for (int r = 0; r < 8; ++r) {
    Cs[hi * 8 + r][col]      = acc0[r] + bcol;
    Cs[16 + hi * 8 + r][col] = acc1[r] + bcol;
  }
  __syncthreads();

  if (tid < 32) {
    float s = 0.0f;
    for (int c = 0; c < 128; ++c) s += Cs[tid][c];
    float mu = s * (1.0f / 128.0f);
    float vs = 0.0f;
    for (int c = 0; c < 128; ++c) { float d = Cs[tid][c] - mu; vs += d * d; }
    mu_s[tid] = mu;
    rs_s[tid] = rsqrtf(vs * (1.0f / 128.0f) + 1e-5f);
  }
  __syncthreads();
  for (int idx = tid; idx < 32 * 32; idx += 256) {   // 4 floats per slot
    int r = idx >> 5, c4 = (idx & 31) * 4;
    float mu = mu_s[r], rs = rs_s[r];
    v4f o;
    #pragma unroll
    for (int q = 0; q < 4; ++q)
      o[q] = (Cs[r][c4 + q] - mu) * rs * gamma[c4 + q] + beta[c4 + q];
    *(v4f*)&out[(row0 + r) * 128 + c4] = o;
  }
}

// ---------------------------------------------------------------------------
// K6: Y_nodes = LN(one-hot gather of yn_w rows). One wave32 per point;
// LN over 128 via wave32 shuffle reduction (4 cols/lane).
// ---------------------------------------------------------------------------
__global__ __launch_bounds__(256) void ynodes_kernel(
    const int* __restrict__ Y_t, const float* __restrict__ yn_w,
    const float* __restrict__ g, const float* __restrict__ bta,
    float* __restrict__ out, long nPts) {
  long p = (long)blockIdx.x * 8 + (threadIdx.x >> 5);
  if (p >= nPts) return;
  const int lane = threadIdx.x & 31;
  const int t = Y_t[p];
  const int gg = G_TBL[t], pd = P_TBL[t];
  float v[4]; float s = 0.0f;
  #pragma unroll
  for (int q = 0; q < 4; ++q) {
    int c = lane * 4 + q;
    v[q] = yn_w[t * 128 + c] + yn_w[(120 + gg) * 128 + c] + yn_w[(139 + pd) * 128 + c];
    s += v[q];
  }
  #pragma unroll
  for (int msk = 16; msk > 0; msk >>= 1) s += __shfl_xor(s, msk, 32);
  float mu = s * (1.0f / 128.0f);
  float vs = 0.0f;
  #pragma unroll
  for (int q = 0; q < 4; ++q) { float d = v[q] - mu; vs += d * d; }
  #pragma unroll
  for (int msk = 16; msk > 0; msk >>= 1) vs += __shfl_xor(vs, msk, 32);
  float rs = rsqrtf(vs * (1.0f / 128.0f) + 1e-5f);
  v4f o;
  #pragma unroll
  for (int q = 0; q < 4; ++q) {
    int c = lane * 4 + q;
    o[q] = (v[q] - mu) * rs * g[c] + bta[c];
  }
  *(v4f*)&out[p * 128 + lane * 4] = o;
}

// ---------------------------------------------------------------------------
extern "C" void kernel_launch(void* const* d_in, const int* in_sizes, int n_in,
                              void* d_out, int out_size, void* d_ws, size_t ws_size,
                              hipStream_t stream) {
  const float* X      = (const float*)d_in[0];
  const float* mask   = (const float*)d_in[1];
  const int*   R_idx  = (const int*)d_in[2];
  const int*   chains = (const int*)d_in[3];
  const float* Y      = (const float*)d_in[4];
  const float* Y_m    = (const float*)d_in[5];
  const int*   Y_t    = (const int*)d_in[6];
  const float* pe_w   = (const float*)d_in[7];
  const float* pe_b   = (const float*)d_in[8];
  const float* ee_w   = (const float*)d_in[9];
  const float* ne_g   = (const float*)d_in[10];
  const float* ne_b   = (const float*)d_in[11];
  const float* npd_w  = (const float*)d_in[12];
  const float* npd_b  = (const float*)d_in[13];
  const float* nn_g   = (const float*)d_in[14];
  const float* nn_b   = (const float*)d_in[15];
  const float* tl_w   = (const float*)d_in[16];
  const float* tl_b   = (const float*)d_in[17];
  const float* yn_w   = (const float*)d_in[18];
  const float* ye_w   = (const float*)d_in[19];
  const float* nye_g  = (const float*)d_in[20];
  const float* nye_b  = (const float*)d_in[21];
  const float* nyn_g  = (const float*)d_in[22];
  const float* nyn_b  = (const float*)d_in[23];

  const long BL   = (long)BDIM * LDIM;        // 2048 residues
  const long nE   = BL * KNBR;                // 61440 edges
  const long nP   = BL * MPTS;                // 32768 ligand points
  const long nYER = nP * MPTS;                // 524288 Y-Y edges

  // workspace carve (256B aligned)
  char* wsp = (char*)d_ws;
  auto carve = [&](size_t bytes) {
    char* p = wsp;
    wsp += (bytes + 255) & ~(size_t)255;
    return p;
  };
  float*    atoms  = (float*)carve(BL * 15 * sizeof(float));
  int*      eidx   = (int*)carve(nE * sizeof(int));
  float*    dneigh = (float*)carve(nE * sizeof(float));
  _Float16* featE  = (_Float16*)carve((size_t)nE * 416 * sizeof(_Float16));
  _Float16* featV  = (_Float16*)carve((size_t)nP * 160 * sizeof(_Float16));
  _Float16* featYE = (_Float16*)carve((size_t)nYER * 32 * sizeof(_Float16));
  _Float16* wpE    = (_Float16*)carve((size_t)13 * 4096 * sizeof(_Float16));
  _Float16* wpV    = (_Float16*)carve((size_t)5 * 4096 * sizeof(_Float16));
  _Float16* wpYE   = (_Float16*)carve((size_t)1 * 4096 * sizeof(_Float16));

  // output layout (floats, return order)
  float* out     = (float*)d_out;
  float* outV    = out;                    // 4,194,304
  float* outE    = outV + 4194304;         // 7,864,320
  float* outEidx = outE + 7864320;         //    61,440
  float* outYN   = outEidx + 61440;        // 4,194,304
  float* outYE   = outYN + 4194304;        // 67,108,864
  float* outYM   = outYE + 67108864;       //    32,768

  // weight repacks (tiny, up front)
  repack_w_kernel<<<(13 * 4096 + 255) / 256, 256, 0, stream>>>(ee_w, wpE, 416, 416);
  repack_w_kernel<<<(5 * 4096 + 255) / 256, 256, 0, stream>>>(npd_w, wpV, 148, 160);
  repack_w_kernel<<<(1 * 4096 + 255) / 256, 256, 0, stream>>>(ye_w, wpYE, 16, 32);

  atoms_kernel<<<(int)((BL + 255) / 256), 256, 0, stream>>>(X, atoms, (int)BL);
  dist_topk_kernel<<<(int)BL, 256, 0, stream>>>(X, mask, dneigh, eidx, outEidx);
  edge_feat_kernel<<<(int)((nE + 7) / 8), 256, 0, stream>>>(
      atoms, eidx, dneigh, R_idx, chains, pe_w, pe_b, featE, nE);
  // E = LN(feat416 @ ee_w)      : 1920 blocks, 13 k-steps x 2 WMMA
  wmma_gemm_ln_kernel<416><<<(int)(nE / 32), 256, 0, stream>>>(
      featE, wpE, nullptr, ne_g, ne_b, outE);

  node_feat_kernel<<<(int)((nP + 7) / 8), 256, 0, stream>>>(
      atoms, Y, Y_t, tl_w, tl_b, featV, nP);
  // V = LN(feat148 @ npd_w + b) : 1024 blocks, 5 k-steps x 2 WMMA
  wmma_gemm_ln_kernel<160><<<(int)(nP / 32), 256, 0, stream>>>(
      featV, wpV, npd_b, nn_g, nn_b, outV);

  yedge_feat_kernel<<<(int)((nYER * 32 + 255) / 256), 256, 0, stream>>>(Y, featYE, nYER);
  // Y_edges = LN(rbf16 @ ye_w)  : 16384 blocks, 1 k-step x 2 WMMA
  wmma_gemm_ln_kernel<32><<<(int)(nYER / 32), 256, 0, stream>>>(
      featYE, wpYE, nullptr, nye_g, nye_b, outYE);

  ynodes_kernel<<<(int)((nP + 7) / 8), 256, 0, stream>>>(
      Y_t, yn_w, nyn_g, nyn_b, outYN, nP);

  hipMemcpyAsync(outYM, Y_m, (size_t)nP * sizeof(float),
                 hipMemcpyDeviceToDevice, stream);
}
